// BaselineWithAttention_1494648619252
// MI455X (gfx1250) — matrix operand order repaired and checked by reading, not compiled, not run on hardware
//
#include <hip/hip_runtime.h>
#include <hip/hip_bf16.h>
#include <stdint.h>

// Problem dimensions (fixed by the reference)
#define VOCAB 50257
#define EMBD  512
#define NCLS  4
#define BATCH 32
#define SEQ   2048

// Flash-attention tiling
#define BM   128            // query rows per workgroup (8 waves x 16)
#define BN   32             // kv tile (matches WMMA K=32 for the PV matmul)
#define KCH  (EMBD / 32)    // 16 k-chunks for S = Q K^T
#define DCH  (EMBD / 16)    // 32 output column chunks for acc += P V
#define NKV  (SEQ / BN)     // 64 kv tiles

// Padded LDS row pitches (bank-conflict avoidance, 16B alignment kept)
#define KROWP (EMBD + 8)    // 520 bf16 = 1040B: 4-bank rotation per key row
#define VTP   (BN + 8)      // 40 bf16 = 80B: 20-bank rotation per d row

typedef __bf16 bf16;
typedef __attribute__((ext_vector_type(16))) __bf16 v16bf;
typedef __attribute__((ext_vector_type(8)))  __bf16 v8bf;
typedef __attribute__((ext_vector_type(2)))  __bf16 v2bf;
typedef __attribute__((ext_vector_type(8)))  float  v8f;
typedef __attribute__((ext_vector_type(4)))  unsigned int u32x4;
typedef __attribute__((ext_vector_type(8)))  int    i32x8;
typedef __attribute__((ext_vector_type(4)))  int    i32x4;

#define LOG2E 1.4426950408889634f

// ---- Tensor Data Mover availability / arity detection --------------------
#if defined(__has_builtin)
# if __has_builtin(__builtin_amdgcn_tensor_load_to_lds)
#  define HAVE_TDM 1
# endif
#endif
#ifndef HAVE_TDM
# define HAVE_TDM 0
#endif
#if __has_include(<hip/amd_detail/amd_gfx1250_TDM.h>)
# define TDM_ARITY6 1   // therock-10.0 headers -> 6-arg builtin
#else
# define TDM_ARITY6 0   // ROCm 7.2 -> 5-arg builtin
#endif

// Order-preserving float -> u32 encoding so max-pool can use integer atomicMax.
__device__ __forceinline__ unsigned enc_ord(float f) {
  unsigned u = __float_as_uint(f);
  return (u & 0x80000000u) ? ~u : (u | 0x80000000u);
}
__device__ __forceinline__ float dec_ord(unsigned e) {
  return (e & 0x80000000u) ? __uint_as_float(e & 0x7FFFFFFFu)
                           : __uint_as_float(~e);
}
__device__ __forceinline__ v16bf cat8(v8bf lo, v8bf hi) {
  v16bf r;
#pragma unroll
  for (int i = 0; i < 8; ++i) { r[i] = lo[i]; r[8 + i] = hi[i]; }
  return r;
}

#if HAVE_TDM
// Issue a TDM 2D tile load: 32 rows x 256 dwords from global H into LDS,
// writing the row-padded layout (pad 4 dwords after every 256 dwords).
__device__ __forceinline__ void tdm_issue_tile(uint32_t lds_off, uint64_t gaddr) {
  u32x4 g0;
  g0[0] = 1u;                                                   // count=1, user D#
  g0[1] = lds_off;                                              // lds_addr
  g0[2] = (uint32_t)gaddr;                                      // global_addr[31:0]
  g0[3] = (uint32_t)((gaddr >> 32) & 0x01FFFFFFu) | (2u << 30); // addr[56:32]|type=2
  i32x8 g1;
  g1[0] = (int)((2u << 16)      // data_size = 4B units
              | (1u << 20)      // pad_enable
              | (7u << 22)      // pad_interval: every 256 dwords (one row)
              | (3u << 25));    // pad_amount: 4 dwords (16B)
  g1[1] = (int)((EMBD / 2) << 16);      // tensor_dim0 = 256 dwords / row
  g1[2] = (int)((uint32_t)SEQ << 16);   // tensor_dim1 = 2048 rows
  g1[3] = (int)((EMBD / 2) << 16);      // tile_dim0 = 256 dwords (full row)
  g1[4] = (int)BN;                      // tile_dim1 = 32 rows, tile_dim2 = 0
  g1[5] = (int)(EMBD / 2);              // tensor_dim0_stride = 256 dwords
  g1[6] = 0;
  g1[7] = 0;
  i32x4 z4 = {0, 0, 0, 0};
#if TDM_ARITY6
  i32x8 z8 = {0, 0, 0, 0, 0, 0, 0, 0};
  __builtin_amdgcn_tensor_load_to_lds(g0, g1, z4, z4, z8, 0);
#else
  __builtin_amdgcn_tensor_load_to_lds(g0, g1, z4, z4, 0);
#endif
}
#endif

// ---------------------------------------------------------------------------
// Kernel 0: init pooled max accumulators (0 is the smallest value in enc_ord order)
// ---------------------------------------------------------------------------
__global__ void init_pooled_kernel(unsigned* __restrict__ pooled) {
  int i = blockIdx.x * blockDim.x + threadIdx.x;
  if (i < BATCH * EMBD) pooled[i] = 0u;
}

// ---------------------------------------------------------------------------
// Kernel 1: H[b,n,:] = bf16(emb[x[b,n],:])   (one row per block)
// ---------------------------------------------------------------------------
__global__ __launch_bounds__(256) void gather_embed_kernel(
    const int* __restrict__ x, const float* __restrict__ emb,
    bf16* __restrict__ H) {
  const int row = blockIdx.x;                 // b*SEQ + n
  const int tok = x[row];
  const float* src = emb + (size_t)tok * EMBD;
  bf16* dst = H + (size_t)row * EMBD;
  const int d = threadIdx.x * 2;
  const float2 v = *(const float2*)(src + d);
  v2bf o; o[0] = (bf16)v.x; o[1] = (bf16)v.y;
  *(v2bf*)(dst + d) = o;
}

// ---------------------------------------------------------------------------
// Kernel 2: flash attention + fused max-pool.
// grid = (SEQ/BM, BATCH), 256 threads (8 waves), wave w owns Q rows
// [blockIdx.x*BM + 16w, +16).
// Dynamic LDS: Krow[2] (2 x 33280B, TDM double buffer) | Vt 40960B | P 10240B.
// ---------------------------------------------------------------------------
__global__ __launch_bounds__(256) void attn_flash_kernel(
    const bf16* __restrict__ H, unsigned* __restrict__ pooled) {
  extern __shared__ char smem[];
  bf16* Krow0 = (bf16*)smem;                             // [BN][KROWP]
  bf16* Krow1 = (bf16*)(smem + BN * KROWP * 2);          // [BN][KROWP]
  bf16* Vt    = (bf16*)(smem + 2 * BN * KROWP * 2);      // [EMBD][VTP]
  bf16* Pb    = (bf16*)(smem + 2 * BN * KROWP * 2 + EMBD * VTP * 2); // [8][16][VTP]

  const int tid  = threadIdx.x;
  const int wave = tid >> 5;
  const int lane = tid & 31;
  const int l16  = lane & 15;
  const int lh   = lane >> 4;            // lane half: selects K-range of frags
  const int b    = blockIdx.y;
  const int q0   = blockIdx.x * BM + wave * 16;

  const bf16* Hb = H + (size_t)b * SEQ * EMBD;

  // ---- Q strip -> 16 A-fragments, kept in VGPRs for the whole kernel ----
  // A frags (16x32), contiguous k order chosen here: lanes 0-15 hold K 0..15, lanes 16-31
  // hold K 16..31; M = lane%16 on both halves.
  v16bf Aq[KCH];
  {
    const bf16* qp = Hb + (size_t)(q0 + l16) * EMBD;
#pragma unroll
    for (int kc = 0; kc < KCH; ++kc) {
      const int kb = kc * 32 + lh * 16;
      Aq[kc] = cat8(*(const v8bf*)(qp + kb), *(const v8bf*)(qp + kb + 8));
    }
  }

  // Flash-attention state: 32 f32 accumulator fragments + per-row stats.
  // C/D layout: VGPR r holds row r (lanes 0-15) / row r+8 (lanes 16-31),
  // column = lane%16 -- identical indexing for acc, mrow, lrow.
  v8f acc[DCH];
#pragma unroll
  for (int n = 0; n < DCH; ++n)
#pragma unroll
    for (int r = 0; r < 8; ++r) acc[n][r] = 0.0f;

  float mrow[8], lrow[8];
#pragma unroll
  for (int r = 0; r < 8; ++r) { mrow[r] = -3.0e38f; lrow[r] = 0.0f; }

  bf16* pw = Pb + wave * (16 * VTP);

#if HAVE_TDM
  // Prologue: kick off the DMA for tile 0 into buffer 0.
  if (wave == 0)
    tdm_issue_tile((uint32_t)(uintptr_t)Krow0, (uint64_t)(uintptr_t)Hb);
#endif

  for (int kv = 0; kv < NKV; ++kv) {
    const int kv0 = kv * BN;
    bf16* KrowCur = (kv & 1) ? Krow1 : Krow0;

#if HAVE_TDM
    // Wait for this tile's DMA (issuing wave only), then make it visible.
    if (wave == 0) __builtin_amdgcn_s_wait_tensorcnt(0);
    __syncthreads();
    // Immediately start the DMA for the *next* tile into the other buffer:
    // the barrier above guarantees every wave finished reading it (it was the
    // compute buffer in the previous iteration).
    if (wave == 0 && kv + 1 < NKV) {
      bf16* KrowNxt = (kv & 1) ? Krow0 : Krow1;
      tdm_issue_tile((uint32_t)(uintptr_t)KrowNxt,
                     (uint64_t)(uintptr_t)(Hb + (size_t)(kv0 + BN) * EMBD));
    }
#else
    __syncthreads();  // previous iteration's readers done before restaging
    {
      const bf16* src = Hb + (size_t)kv0 * EMBD;
#pragma unroll
      for (int i = 0; i < (BN * EMBD / 8) / 256; ++i) {
        const int idx = tid + i * 256;
        const int row = idx >> 6;           // EMBD/8 = 64 octets per row
        const int c8  = (idx & 63) * 8;
        *(v8bf*)(KrowCur + (size_t)row * KROWP + c8) =
            *(const v8bf*)(src + (size_t)row * EMBD + c8);
      }
    }
    __syncthreads();
#endif

    // ---- build Vt (transposed tile) for PV B-fragments ----
#pragma unroll
    for (int i = 0; i < (BN * EMBD / 8) / 256; ++i) {
      const int idx = tid + i * 256;
      const int key = idx >> 6;
      const int d0  = (idx & 63) * 8;
      v8bf v = *(const v8bf*)(KrowCur + (size_t)key * KROWP + d0);
#pragma unroll
      for (int e = 0; e < 8; ++e) Vt[(d0 + e) * VTP + key] = v[e];
    }
    __syncthreads();

    // prefetch the tile after next into cache while we compute on this one
    if (kv + 2 < NKV)
      __builtin_prefetch(Hb + (size_t)(kv0 + 2 * BN) * EMBD + tid * 64, 0, 1);

    // ---- S = Q K^T : two 16x16 score fragments, K-dim = EMBD ----
    // B frags (32x16), same contiguous k order as Aq: lanes 0-15 hold K 0..15,
    // lanes 16-31 hold K 16..31; N = lane%16 = key index.
    v8f S0, S1;
#pragma unroll
    for (int r = 0; r < 8; ++r) { S0[r] = 0.0f; S1[r] = 0.0f; }
    {
      const bf16* kp0 = KrowCur + (size_t)l16 * KROWP;         // keys 0..15
      const bf16* kp1 = KrowCur + (size_t)(16 + l16) * KROWP;  // keys 16..31
#pragma unroll
      for (int kc = 0; kc < KCH; ++kc) {
        const int kb = kc * 32 + lh * 16;
        v16bf B0 = *(const v16bf*)(kp0 + kb);
        v16bf B1 = *(const v16bf*)(kp1 + kb);
        S0 = __builtin_amdgcn_wmma_f32_16x16x32_bf16(false, Aq[kc], false, B0,
                                                     (short)0, S0, false, false);
        S1 = __builtin_amdgcn_wmma_f32_16x16x32_bf16(false, Aq[kc], false, B1,
                                                     (short)0, S1, false, false);
      }
    }

    // ---- online softmax (row reductions across the 16-lane halves) ----
    float corr[8];
#pragma unroll
    for (int r = 0; r < 8; ++r) {
      float v = fmaxf(S0[r], S1[r]);
      v = fmaxf(v, __shfl_xor(v, 1));
      v = fmaxf(v, __shfl_xor(v, 2));
      v = fmaxf(v, __shfl_xor(v, 4));
      v = fmaxf(v, __shfl_xor(v, 8));
      const float mnew = fmaxf(mrow[r], v);
      corr[r] = exp2f((mrow[r] - mnew) * LOG2E);
      mrow[r] = mnew;
      S0[r] = exp2f((S0[r] - mnew) * LOG2E);
      S1[r] = exp2f((S1[r] - mnew) * LOG2E);
      float s = S0[r] + S1[r];
      s += __shfl_xor(s, 1);
      s += __shfl_xor(s, 2);
      s += __shfl_xor(s, 4);
      s += __shfl_xor(s, 8);
      lrow[r] = lrow[r] * corr[r] + s;
    }
#pragma unroll
    for (int n = 0; n < DCH; ++n)
#pragma unroll
      for (int r = 0; r < 8; ++r) acc[n][r] *= corr[r];

    // ---- P: C-layout -> A-fragment via per-wave LDS round-trip ----
#pragma unroll
    for (int r = 0; r < 8; ++r) {
      pw[(r + 8 * lh) * VTP + l16]      = (bf16)S0[r];
      pw[(r + 8 * lh) * VTP + 16 + l16] = (bf16)S1[r];
    }
    asm volatile("s_wait_dscnt 0" ::: "memory");  // per-wave buffer, RAW only
    v16bf Pf;
    {
      const bf16* pr = pw + (size_t)l16 * VTP + lh * 16;
      Pf = cat8(*(const v8bf*)pr, *(const v8bf*)(pr + 8));
    }

    // ---- acc += P @ V over all 32 output column chunks ----
#pragma unroll
    for (int n = 0; n < DCH; ++n) {
      const bf16* vp = Vt + (size_t)(n * 16 + l16) * VTP + lh * 16;
      v16bf Vf = *(const v16bf*)vp;
      acc[n] = __builtin_amdgcn_wmma_f32_16x16x32_bf16(false, Pf, false, Vf,
                                                       (short)0, acc[n], false, false);
    }
  }

  // ---- finalize: out = acc / l, fold max over this wave's 16 rows, atomic ----
  float linv[8];
#pragma unroll
  for (int r = 0; r < 8; ++r) linv[r] = 1.0f / lrow[r];
#pragma unroll
  for (int n = 0; n < DCH; ++n) {
    float v = -3.0e38f;
#pragma unroll
    for (int r = 0; r < 8; ++r) v = fmaxf(v, acc[n][r] * linv[r]);
    v = fmaxf(v, __shfl_xor(v, 16));  // merge rows 0-7 with rows 8-15
    if (lh == 0)
      atomicMax(&pooled[b * EMBD + n * 16 + l16], enc_ord(v));
  }
}

// ---------------------------------------------------------------------------
// Kernel 3: logits[b,c] = pooled[b,:] . W[c,:] + bias[c]   (tiny)
// ---------------------------------------------------------------------------
__global__ __launch_bounds__(128) void logits_kernel(
    const unsigned* __restrict__ pooled, const float* __restrict__ W,
    const float* __restrict__ bias, float* __restrict__ out) {
  const int t = threadIdx.x;            // 128 = 32 batches x 4 classes
  const int bb = t >> 2, c = t & 3;
  float s = bias[c];
  for (int d = 0; d < EMBD; ++d)
    s = fmaf(dec_ord(pooled[bb * EMBD + d]), W[c * EMBD + d], s);
  out[t] = s;
}

// ---------------------------------------------------------------------------
extern "C" void kernel_launch(void* const* d_in, const int* in_sizes, int n_in,
                              void* d_out, int out_size, void* d_ws, size_t ws_size,
                              hipStream_t stream) {
  (void)in_sizes; (void)n_in; (void)out_size; (void)ws_size;
  const int*   x   = (const int*)d_in[0];
  const float* emb = (const float*)d_in[1];
  const float* W   = (const float*)d_in[2];
  const float* bia = (const float*)d_in[3];
  float* out = (float*)d_out;

  // Workspace layout: H (bf16, 64MB, L2-resident) | pooled (u32, 64KB)
  bf16* H = (bf16*)d_ws;
  unsigned* pooled =
      (unsigned*)((char*)d_ws + (size_t)BATCH * SEQ * EMBD * sizeof(bf16));

  init_pooled_kernel<<<(BATCH * EMBD + 255) / 256, 256, 0, stream>>>(pooled);
  gather_embed_kernel<<<BATCH * SEQ, 256, 0, stream>>>(x, emb, H);

  dim3 grid(SEQ / BM, BATCH);
  const size_t lds_bytes =
      (size_t)(2 * BN * KROWP * 2 + EMBD * VTP * 2 + 8 * 16 * VTP * 2); // 117760
  attn_flash_kernel<<<grid, 256, lds_bytes, stream>>>(H, pooled);

  logits_kernel<<<1, 128, 0, stream>>>(pooled, W, bia, out);
}
